// Encoder_19198503813777
// MI455X (gfx1250) — matrix operand order repaired and checked by reading, not compiled, not run on hardware
//
#include <hip/hip_runtime.h>

typedef __attribute__((ext_vector_type(16))) __bf16 v16bf;
typedef __attribute__((ext_vector_type(8)))  float  v8f;
typedef long long i64;

#define FDIM 512
#define HID  32

// ---------------------------------------------------------------- utilities
__global__ void zero_f32(float* __restrict__ p, long long n) {
  long long i = (long long)blockIdx.x * blockDim.x + threadIdx.x;
  long long stride = (long long)gridDim.x * blockDim.x;
  for (; i < n; i += stride) p[i] = 0.f;
}

// Fold feature masks into W1 and transpose both weights into bf16 [out][in]
// W1t: [3][HID][FDIM] bf16 ; W2t: [HID][HID] bf16
__global__ void prep_weights(const float* __restrict__ W1,
                             const float* __restrict__ fm1,
                             const float* __restrict__ fm2,
                             const float* __restrict__ W2,
                             unsigned short* __restrict__ W1t_u,
                             unsigned short* __restrict__ W2t_u) {
  __bf16* W1t = (__bf16*)W1t_u;
  __bf16* W2t = (__bf16*)W2t_u;
  int idx = blockIdx.x * blockDim.x + threadIdx.x;
  const int tot1 = 3 * HID * FDIM;
  if (idx < tot1) {
    int v = idx / (HID * FDIM);
    int r = idx % (HID * FDIM);
    int n = r / FDIM;
    int k = r % FDIM;
    float m = (v == 0) ? 1.f : ((v == 1) ? fm1[k] : fm2[k]);
    W1t[idx] = (__bf16)(W1[k * HID + n] * m);
  } else if (idx < tot1 + HID * HID) {
    int r = idx - tot1;
    int n = r / HID, k = r % HID;
    W2t[r] = (__bf16)(W2[k * HID + n]);
  }
}

// Per-view weighted degree via atomics (deg buffer later becomes dinv in place)
__global__ void deg_accum(const i64* __restrict__ ei, const float* __restrict__ ew,
                          const float* __restrict__ em1, const float* __restrict__ em2,
                          float* __restrict__ deg, long long E, int N) {
  long long e = (long long)blockIdx.x * blockDim.x + threadIdx.x;
  if (e >= E) return;
  int dst = (int)ei[E + e];
  float w = ew[e];
  atomicAdd(&deg[dst], w);
  atomicAdd(&deg[N + dst], w * em1[e]);
  atomicAdd(&deg[2 * N + dst], w * em2[e]);
}

__global__ void finish_dinv(float* __restrict__ dinv, int n) {
  int i = blockIdx.x * blockDim.x + threadIdx.x;
  if (i < n) dinv[i] = rsqrtf(dinv[i] + 1.0f);  // +1 self-loop, always finite
}

// -------------------------------------------------------- GEMM1: x @ W1t_v
// One pass over x (205MB) computes all 3 views. 1024 thr = 32 waves, each wave
// owns a 16-row strip and 6 fp32 accumulators (3 views x 2 n-tiles).
// Software-pipelined: A-loads for k0+32 issue before the k0 WMMA batch; all 6
// B fragments are fetched with one DS drain, then 6 WMMAs go back-to-back.
__global__ void gemm1_wmma(const float* __restrict__ x,
                           const unsigned short* __restrict__ W1t_u,
                           float* __restrict__ h1, int N) {
  extern __shared__ v16bf smem_dyn[];            // 32B-aligned base
  __bf16* sW = (__bf16*)smem_dyn;                // [3][HID][FDIM]
  const int SWU = (3 * HID * FDIM) / 2;          // as dwords
  {
    const unsigned int* gsrc = (const unsigned int*)W1t_u;
    unsigned int* ldst = (unsigned int*)sW;
    for (int i = threadIdx.x; i < SWU; i += blockDim.x) ldst[i] = gsrc[i];
  }
  __syncthreads();

  int wave = threadIdx.x >> 5;
  int lane = threadIdx.x & 31;
  int m0 = (blockIdx.x * (blockDim.x >> 5) + wave) * 16;
  if (m0 >= N) return;                           // wave-uniform: EXEC stays full

  int half = lane >> 4;                          // lane group 0/1
  int l16  = lane & 15;
  int row  = m0 + l16;                           // A-fragment row for this lane
  int kA   = half * 16;                          // contiguous k = 16h + s (matches B)
  int kB   = half * 16;                          // contiguous k = 16h + s (matches A)

  // per-tile B base pointers (hoists address math out of the WMMA cluster)
  const __bf16* bbase[6];
#pragma unroll
  for (int v = 0; v < 3; ++v)
#pragma unroll
    for (int nt = 0; nt < 2; ++nt)
      bbase[v * 2 + nt] = sW + ((v * HID + nt * 16 + l16) * FDIM + kB);

  v8f acc[6] = {};                               // [view][ntile]
  const float* xrow = x + (long long)row * FDIM;

  // prologue: A data for k0 = 0
  float4 c0 = ((const float4*)(xrow + kA))[0];
  float4 c1 = ((const float4*)(xrow + kA))[1];
  float4 c2 = ((const float4*)(xrow + 8 + kA))[0];
  float4 c3 = ((const float4*)(xrow + 8 + kA))[1];

  for (int k0 = 0; k0 < FDIM; k0 += 32) {
    // convert current A chunk to the bf16 fragment
    v16bf a;
    a[0]=(__bf16)c0.x;  a[1]=(__bf16)c0.y;  a[2]=(__bf16)c0.z;  a[3]=(__bf16)c0.w;
    a[4]=(__bf16)c1.x;  a[5]=(__bf16)c1.y;  a[6]=(__bf16)c1.z;  a[7]=(__bf16)c1.w;
    a[8]=(__bf16)c2.x;  a[9]=(__bf16)c2.y;  a[10]=(__bf16)c2.z; a[11]=(__bf16)c2.w;
    a[12]=(__bf16)c3.x; a[13]=(__bf16)c3.y; a[14]=(__bf16)c3.z; a[15]=(__bf16)c3.w;

    // issue next k-step's global loads early (latency hidden under DS + WMMA)
    if (k0 + 32 < FDIM) {
      const float* nx = xrow + k0 + 32;
      c0 = ((const float4*)(nx + kA))[0];
      c1 = ((const float4*)(nx + kA))[1];
      c2 = ((const float4*)(nx + 8 + kA))[0];
      c3 = ((const float4*)(nx + 8 + kA))[1];
    }

    // fetch all 6 B fragments, then run the 6 WMMAs back-to-back
    v16bf bf[6];
#pragma unroll
    for (int t = 0; t < 6; ++t)
      bf[t] = *(const v16bf*)(bbase[t] + k0);
#pragma unroll
    for (int t = 0; t < 6; ++t)
      acc[t] = __builtin_amdgcn_wmma_f32_16x16x32_bf16(
          false, a, false, bf[t], (short)0, acc[t], false, false);
  }

#pragma unroll
  for (int v = 0; v < 3; ++v)
#pragma unroll
    for (int nt = 0; nt < 2; ++nt) {
      float* out = h1 + ((long long)v * N + m0) * HID + nt * 16 + l16;
#pragma unroll
      for (int j = 0; j < 8; ++j)
        out[(long long)(j + half * 8) * HID] = acc[v * 2 + nt][j];
    }
}

// -------------------------------------------------------- GEMM2: g_v @ W2t
__global__ void gemm2_wmma(const float* __restrict__ g,
                           const unsigned short* __restrict__ W2t_u,
                           float* __restrict__ z, int N) {
  __shared__ __align__(32) __bf16 sW[HID * HID];
  for (int i = threadIdx.x; i < (HID * HID) / 2; i += blockDim.x)
    ((unsigned int*)sW)[i] = ((const unsigned int*)W2t_u)[i];
  __syncthreads();

  int wave = threadIdx.x >> 5;
  int lane = threadIdx.x & 31;
  int m0 = (blockIdx.x * (blockDim.x >> 5) + wave) * 16;
  if (m0 >= N) return;

  int half = lane >> 4;
  int l16  = lane & 15;
  int row  = m0 + l16;
  int kA   = half * 16;
  int kB   = half * 16;

  v16bf bfrag[2];
#pragma unroll
  for (int nt = 0; nt < 2; ++nt)
    bfrag[nt] = *(const v16bf*)(sW + (nt * 16 + l16) * HID + kB);

  v8f acc[6] = {};
#pragma unroll
  for (int v = 0; v < 3; ++v) {
    const float* grow = g + ((long long)v * N + row) * HID;
    const float4* p1 = (const float4*)(grow + kA);
    const float4* p2 = (const float4*)(grow + 8 + kA);
    float4 c0 = p1[0], c1 = p1[1], c2 = p2[0], c3 = p2[1];
    v16bf a;
    a[0]=(__bf16)c0.x;  a[1]=(__bf16)c0.y;  a[2]=(__bf16)c0.z;  a[3]=(__bf16)c0.w;
    a[4]=(__bf16)c1.x;  a[5]=(__bf16)c1.y;  a[6]=(__bf16)c1.z;  a[7]=(__bf16)c1.w;
    a[8]=(__bf16)c2.x;  a[9]=(__bf16)c2.y;  a[10]=(__bf16)c2.z; a[11]=(__bf16)c2.w;
    a[12]=(__bf16)c3.x; a[13]=(__bf16)c3.y; a[14]=(__bf16)c3.z; a[15]=(__bf16)c3.w;
#pragma unroll
    for (int nt = 0; nt < 2; ++nt)
      acc[v * 2 + nt] = __builtin_amdgcn_wmma_f32_16x16x32_bf16(
          false, a, false, bfrag[nt], (short)0, acc[v * 2 + nt], false, false);
  }

#pragma unroll
  for (int v = 0; v < 3; ++v)
#pragma unroll
    for (int nt = 0; nt < 2; ++nt) {
      float* out = z + ((long long)v * N + m0) * HID + nt * 16 + l16;
#pragma unroll
      for (int j = 0; j < 8; ++j)
        out[(long long)(j + half * 8) * HID] = acc[v * 2 + nt][j];
    }
}

// ------------------------------------------------- edge-parallel aggregation
// One wave per edge, lane == feature (HID==32==wave32). h arrays are L2-resident
// (12.8MB/view), so gathers + atomics stay on-chip; edge_index streams from HBM.
__global__ void edge_agg(const i64* __restrict__ ei, const float* __restrict__ ew,
                         const float* __restrict__ em1, const float* __restrict__ em2,
                         const float* __restrict__ dinv, const float* __restrict__ h,
                         float* __restrict__ agg, long long E, int N) {
  long long w = ((long long)blockIdx.x * blockDim.x + threadIdx.x) >> 5;
  int lane = threadIdx.x & 31;
  if (w >= E) return;
  int src = (int)ei[w];
  int dst = (int)ei[E + w];
  float we = ew[w];
  float wv[3] = {we, we * em1[w], we * em2[w]};
#pragma unroll
  for (int v = 0; v < 3; ++v) {
    if (wv[v] != 0.f) {                       // wave-uniform branch
      float norm = dinv[v * N + src] * wv[v] * dinv[v * N + dst];
      float val = h[((long long)v * N + src) * HID + lane] * norm;
      atomicAdd(&agg[((long long)v * N + dst) * HID + lane], val);
    }
  }
}

// out = relu(agg + h*dinv^2 + bias)
__global__ void selfloop_bias_relu(const float* __restrict__ agg,
                                   const float* __restrict__ h,
                                   const float* __restrict__ dinv,
                                   const float* __restrict__ bias,
                                   float* __restrict__ out, int N) {
  int idx = blockIdx.x * blockDim.x + threadIdx.x;
  int tot = 3 * N * HID;
  if (idx >= tot) return;
  int v = idx / (N * HID);
  int r = idx % (N * HID);
  int i = r / HID;
  int f = r % HID;
  float di = dinv[v * N + i];
  float val = agg[idx] + h[idx] * di * di + bias[f];
  out[idx] = fmaxf(val, 0.f);
}

// ----------------------------------------------------------------- launcher
extern "C" void kernel_launch(void* const* d_in, const int* in_sizes, int n_in,
                              void* d_out, int out_size, void* d_ws, size_t ws_size,
                              hipStream_t stream) {
  const float* x   = (const float*)d_in[0];
  const i64*   ei  = (const i64*)d_in[1];     // int64 edge_index [2][E]
  const float* ew  = (const float*)d_in[2];
  const float* em1 = (const float*)d_in[3];
  const float* em2 = (const float*)d_in[4];
  const float* fm1 = (const float*)d_in[5];
  const float* fm2 = (const float*)d_in[6];
  const float* W1  = (const float*)d_in[7];
  const float* b1  = (const float*)d_in[8];
  const float* W2  = (const float*)d_in[9];
  const float* b2  = (const float*)d_in[10];

  const int N = in_sizes[0] / FDIM;
  const long long E = in_sizes[2];

  // workspace carve-up
  float* ws = (float*)d_ws;
  const size_t nv = (size_t)3 * N;            // dinv / degree
  const size_t nh = (size_t)3 * N * HID;      // per big buffer
  float* dinv = ws;
  float* agg  = dinv + nv;
  float* h1   = agg + nh;
  float* g    = h1 + nh;
  unsigned short* W1t = (unsigned short*)(g + nh);
  unsigned short* W2t = W1t + 3 * HID * FDIM;
  float* out = (float*)d_out;

  const int tot_h = (int)nh;
  const int gemmBlocks = (N + 511) / 512;     // 32 waves * 16 rows per block
  const long long edgeThreads = E * 32;

  // 1) zero degree + agg (contiguous region)
  zero_f32<<<2048, 256, 0, stream>>>(dinv, (long long)(nv + nh));
  // 2) fold masks into bf16 transposed weights
  {
    int tot = 3 * HID * FDIM + HID * HID;
    prep_weights<<<(tot + 255) / 256, 256, 0, stream>>>(W1, fm1, fm2, W2, W1t, W2t);
  }
  // 3) per-view degrees, 4) dinv = rsqrt(deg+1)
  deg_accum<<<(int)((E + 255) / 256), 256, 0, stream>>>(ei, ew, em1, em2, dinv, E, N);
  finish_dinv<<<(int)((nv + 255) / 256), 256, 0, stream>>>(dinv, (int)nv);
  // 5) layer-1 GEMM for all 3 views in one pass over x
  gemm1_wmma<<<gemmBlocks, 1024, 3 * HID * FDIM * sizeof(unsigned short), stream>>>(
      x, W1t, h1, N);
  // 6) layer-1 aggregation, 7) self-loop + bias + relu -> g
  edge_agg<<<(int)((edgeThreads + 255) / 256), 256, 0, stream>>>(
      ei, ew, em1, em2, dinv, h1, agg, E, N);
  selfloop_bias_relu<<<(tot_h + 255) / 256, 256, 0, stream>>>(agg, h1, dinv, b1, g, N);
  // 8) reset agg for layer 2
  zero_f32<<<2048, 256, 0, stream>>>(agg, (long long)nh);
  // 9) layer-2 GEMM (g -> h1 reused as pre-agg), 10) aggregation, 11) final relu
  gemm2_wmma<<<gemmBlocks, 1024, 0, stream>>>(g, W2t, h1, N);
  edge_agg<<<(int)((edgeThreads + 255) / 256), 256, 0, stream>>>(
      ei, ew, em1, em2, dinv, h1, agg, E, N);
  selfloop_bias_relu<<<(tot_h + 255) / 256, 256, 0, stream>>>(agg, h1, dinv, b2, out, N);
}
